// DistributedGymWorldModel_77403900609094
// MI455X (gfx1250) — matrix operand order repaired and checked by reading, not compiled, not run on hardware
//
#include <hip/hip_runtime.h>

// MI455X / gfx1250, wave32. Dense layers via v_wmma_f32_16x16x32_f16 (f32 acc).
// Weights pre-transposed + converted to f16 [N][Kpad] once per launch; all
// activations f16. GEMM K-loop stages tiles with global_load_async_to_lds_b128
// (ASYNCcnt, zero staging VGPRs -> no spill possible), double-buffered LDS:
// 4x async b128 + 12x ds_load_b128 + 8x v_wmma + s_wait_asynccnt per K-tile.

typedef _Float16 h16 __attribute__((ext_vector_type(16)));
typedef float    f8  __attribute__((ext_vector_type(8)));

#define DEV_INLINE __device__ __forceinline__

constexpr int LDSS = 40;          // 32 data halfs + 8 pad; rows stay 16B aligned
constexpr int HBUF = 128 * LDSS;  // one LDS buffer (halfs)

struct U4x2 { uint4 lo, hi; };

// One 16x32 f16 WMMA operand fragment from an LDS tile row.
// Lane L holds row (L%16); halfs [0..7] = K (L/16)*8+0..7,
// halfs [8..15] = K 16+(L/16)*8+0..7 -> two contiguous 16B chunks.
DEV_INLINE h16 load_frag(const _Float16* rowbase, int hsel) {
  U4x2 p;
  p.lo = *reinterpret_cast<const uint4*>(rowbase + hsel * 8);
  p.hi = *reinterpret_cast<const uint4*>(rowbase + 16 + hsel * 8);
  return __builtin_bit_cast(h16, p);
}

// LDS byte address of a __shared__ pointer: flat shared addr truncates to the
// 32-bit LDS offset (aperture scheme, ISA 10.2).
DEV_INLINE unsigned lds_addr(const _Float16* p) {
  return (unsigned)(size_t)p;
}

// 4x global -> LDS async b128 copies (ASYNCcnt; no VGPR staging).
DEV_INLINE void stage_async(const _Float16* pa0, const _Float16* pa1,
                            const _Float16* pb0, const _Float16* pb1,
                            unsigned la0, unsigned la1, unsigned lb0, unsigned lb1) {
  asm volatile(
      "global_load_async_to_lds_b128 %0, %4, off\n\t"
      "global_load_async_to_lds_b128 %1, %5, off\n\t"
      "global_load_async_to_lds_b128 %2, %6, off\n\t"
      "global_load_async_to_lds_b128 %3, %7, off"
      :: "v"(la0), "v"(la1), "v"(lb0), "v"(lb1),
         "v"(pa0), "v"(pa1), "v"(pb0), "v"(pb1)
      : "memory");
}

DEV_INLINE void wait_async() {
  asm volatile("s_wait_asynccnt 0x0" ::: "memory");
}

// 12x ds_load_b128 + 8x v_wmma on one LDS buffer
DEV_INLINE void wmma_tile(const _Float16* aBase, const _Float16* bBase,
                          int hsel, f8 (&acc)[2][4]) {
  h16 af0 = load_frag(aBase, hsel);
  h16 af1 = load_frag(aBase + 16 * LDSS, hsel);
  h16 bf0 = load_frag(bBase, hsel);
  h16 bf1 = load_frag(bBase + 16 * LDSS, hsel);
  h16 bf2 = load_frag(bBase + 32 * LDSS, hsel);
  h16 bf3 = load_frag(bBase + 48 * LDSS, hsel);
  acc[0][0] = __builtin_amdgcn_wmma_f32_16x16x32_f16(false, af0, false, bf0, (short)0, acc[0][0], false, false);
  acc[0][1] = __builtin_amdgcn_wmma_f32_16x16x32_f16(false, af0, false, bf1, (short)0, acc[0][1], false, false);
  acc[0][2] = __builtin_amdgcn_wmma_f32_16x16x32_f16(false, af0, false, bf2, (short)0, acc[0][2], false, false);
  acc[0][3] = __builtin_amdgcn_wmma_f32_16x16x32_f16(false, af0, false, bf3, (short)0, acc[0][3], false, false);
  acc[1][0] = __builtin_amdgcn_wmma_f32_16x16x32_f16(false, af1, false, bf0, (short)0, acc[1][0], false, false);
  acc[1][1] = __builtin_amdgcn_wmma_f32_16x16x32_f16(false, af1, false, bf1, (short)0, acc[1][1], false, false);
  acc[1][2] = __builtin_amdgcn_wmma_f32_16x16x32_f16(false, af1, false, bf2, (short)0, acc[1][2], false, false);
  acc[1][3] = __builtin_amdgcn_wmma_f32_16x16x32_f16(false, af1, false, bf3, (short)0, acc[1][3], false, false);
}

// ---- f32 -> f16 elementwise (activations)
__global__ __launch_bounds__(256) void cvt_f16(const float* __restrict__ src,
                                               _Float16* __restrict__ dst, int n) {
  const int i = blockIdx.x * 256 + threadIdx.x;
  if (i < n) dst[i] = (_Float16)src[i];
}

// ---- one-shot per launch: W f32 [Kc x Nw] -> Wt f16 [Nw x Kpad], zero-pad k>=Kc.
__global__ __launch_bounds__(256) void transpose_w(
    const float* __restrict__ W, int Kc, int Nw,
    _Float16* __restrict__ Wt, int Kpad)
{
  __shared__ _Float16 t[32 * 33];
  const int n0 = blockIdx.x * 32;
  const int k0 = blockIdx.y * 32;
  const int tid = threadIdx.x;
#pragma unroll
  for (int e = 0; e < 4; ++e) {
    const int idx = e * 256 + tid;     // 1024 = 32k x 32n, lanes sweep n
    const int kk = idx >> 5, nn = idx & 31;
    float v = 0.0f;
    if (k0 + kk < Kc) v = W[(size_t)(k0 + kk) * Nw + n0 + nn];
    t[kk * 33 + nn] = (_Float16)v;
  }
  __syncthreads();
#pragma unroll
  for (int e = 0; e < 4; ++e) {
    const int idx = e * 256 + tid;     // lanes sweep k -> contiguous f16 writes
    const int nn = idx >> 5, kk = idx & 31;
    Wt[(size_t)(n0 + nn) * Kpad + k0 + kk] = t[kk * 33 + nn];
  }
}

// C[M x N] = act(A[M x K] @ B[K x N] + bias[N]) (* rowScale[row] if SCALE)
// A : f16 row-major, ld lda; cols [0, KT*32) readable (zero-padded past true K).
// Bt: f16, pre-transposed [N x KT*32] (zero-padded), from transpose_w.
// Grid: (N/128, M/128), block 256 (8 wave32). Wave = 32x64 tile = 2x4 WMMA accs.
template <typename OT, bool RELU, bool SCALE>
__global__ __launch_bounds__(256, 1) void gemm_wmma(
    const _Float16* __restrict__ A, int lda,
    const _Float16* __restrict__ Bt,
    const float* __restrict__ bias,
    OT* __restrict__ C, int ldc,
    const float* __restrict__ rowScale,
    int KT)
{
  __shared__ _Float16 As[2 * HBUF];
  __shared__ _Float16 Bs[2 * HBUF];

  const int ldb  = KT * 32;
  const int m0   = blockIdx.y * 128;
  const int n0   = blockIdx.x * 128;
  const int tid  = threadIdx.x;
  const int lane = tid & 31;
  const int wid  = tid >> 5;
  const int wm   = (wid & 3) * 32;     // 4 waves along M
  const int wn   = (wid >> 2) * 64;    // 2 waves along N
  const int mrow = lane & 15;
  const int hsel = lane >> 4;

  // staging coordinates: 2 chunks of 8 halfs (16B) per thread
  const int r0   = tid >> 2;           // rows 0..63
  const int r1   = r0 + 64;            // rows 64..127
  const int koff = (tid & 3) * 8;

  // per-tile advancing global pointers (step +32 per K-tile)
  const _Float16* pa0 = A  + (size_t)(m0 + r0) * lda + koff;
  const _Float16* pa1 = A  + (size_t)(m0 + r1) * lda + koff;
  const _Float16* pb0 = Bt + (size_t)(n0 + r0) * ldb + koff;
  const _Float16* pb1 = Bt + (size_t)(n0 + r1) * ldb + koff;

  // LDS staging byte addresses (buf0; buf1 via +2*HBUF bytes)
  const unsigned la0 = lds_addr(&As[r0 * LDSS + koff]);
  const unsigned la1 = lds_addr(&As[r1 * LDSS + koff]);
  const unsigned lb0 = lds_addr(&Bs[r0 * LDSS + koff]);
  const unsigned lb1 = lds_addr(&Bs[r1 * LDSS + koff]);

  // fragment base addresses (per wave; buffer via +HBUF halfs)
  const _Float16* fa = &As[(wm + mrow) * LDSS];
  const _Float16* fb = &Bs[(wn + mrow) * LDSS];

  f8 acc[2][4] = {};

  // prologue: tile 0 -> buf0
  stage_async(pa0, pa1, pb0, pb1, la0, la1, lb0, lb1);
  pa0 += 32; pa1 += 32; pb0 += 32; pb1 += 32;
  wait_async();
  __syncthreads();

  int      cm  = 0;                                 // compute buffer (halfs)
  unsigned stB = (unsigned)(HBUF * sizeof(_Float16)); // stage buffer (bytes)
  for (int kt = 0; kt < KT; ++kt) {
    const bool more = (kt + 1) < KT;
    if (more) {                        // async copies fly during the WMMAs below
      stage_async(pa0, pa1, pb0, pb1, la0 + stB, la1 + stB, lb0 + stB, lb1 + stB);
      pa0 += 32; pa1 += 32; pb0 += 32; pb1 += 32;
    }
    wmma_tile(fa + cm, fb + cm, hsel, acc);
    wait_async();                      // staged LDS writes complete
    __syncthreads();
    cm ^= HBUF; stB ^= (unsigned)(2 * HBUF);
  }

  // ---- epilogue. C/D layout: lane L, vgpr r -> M = (L/16)*8 + r, N = L%16
  const int gmbase = m0 + wm + hsel * 8;
  const int gnbase = n0 + wn + (lane & 15);
#pragma unroll
  for (int mi = 0; mi < 2; ++mi) {
#pragma unroll
    for (int ni = 0; ni < 4; ++ni) {
      const int col = gnbase + ni * 16;
      const float bv = bias[col];
      f8 c = acc[mi][ni];
#pragma unroll
      for (int r = 0; r < 8; ++r) {
        const int row = gmbase + mi * 16 + r;
        float v = c[r] + bv;
        if (RELU)  v = fmaxf(v, 0.0f);
        if (SCALE) v *= rowScale[row];
        C[(size_t)row * ldc + col] = (OT)v;
      }
    }
  }
}

// ---- msg_in[row=bi*16384+ai*16+k][0..415] =
//      [z_prev[bi,idx], id[bi,idx], id[bi,ai], edge[ai,k], 0-pad]
__global__ void build_msg_in(const float* __restrict__ z_prev,
                             const float* __restrict__ idf,
                             const int*   __restrict__ nidx,
                             const float* __restrict__ edge,
                             _Float16* __restrict__ msg_in)
{
  const int r   = blockIdx.x;        // 0..131071
  const int bi  = r >> 14;
  const int rem = r & 16383;
  const int ai  = rem >> 4;
  const int k   = rem & 15;
  int idx = nidx[ai * 16 + k]; if (idx < 0) idx = 0;
  const float* sz  = z_prev + ((size_t)bi * 1024 + idx) * 256;
  const float* sid = idf    + ((size_t)bi * 1024 + idx) * 64;
  const float* rid = idf    + ((size_t)bi * 1024 + ai)  * 64;
  const float* ef  = edge   + ((size_t)ai * 16 + k) * 16;
  _Float16* dst = msg_in + (size_t)r * 416;
  for (int c = threadIdx.x; c < 416; c += blockDim.x) {
    float v;
    if      (c < 256) v = sz[c];
    else if (c < 320) v = sid[c - 256];
    else if (c < 384) v = rid[c - 320];
    else if (c < 400) v = ef[c - 384];
    else              v = 0.0f;
    dst[c] = (_Float16)v;
  }
}

// ---- masked mean over K neighbors -> update_in cols [512,768)
__global__ void aggregate_msgs(const _Float16* __restrict__ msg,
                               const float* __restrict__ nmask,
                               _Float16* __restrict__ upd)
{
  const int row = blockIdx.x;        // 0..8191  (= bi*1024+ai)
  const int ai  = row & 1023;
  const int c   = threadIdx.x;       // 256
  const _Float16* m = msg + (size_t)row * 16 * 256;
  float s = 0.0f, d = 0.0f;
#pragma unroll
  for (int k = 0; k < 16; ++k) {
    const float w = nmask[ai * 16 + k];
    s += w * (float)m[k * 256 + c];
    d += w;
  }
  d = fmaxf(d, 1.0f);
  upd[(size_t)row * 864 + 512 + c] = (_Float16)(s / d);
}

// ---- update_in cols [768,864): id_features(64) | obs_mask(1) | zero pad(31)
__global__ void fill_tail(const float* __restrict__ idf,
                          const float* __restrict__ mask,
                          _Float16* __restrict__ upd)
{
  const int row = blockIdx.x;
  const int t = threadIdx.x;         // 96
  float v;
  if      (t < 64)  v = idf[(size_t)row * 64 + t];
  else if (t == 64) v = mask[row];
  else              v = 0.0f;
  upd[(size_t)row * 864 + 768 + t] = (_Float16)v;
}

// ---- GRU pointwise: z_next (f32 out) + f16 copy for the heads
__global__ void gru_pointwise(const float* __restrict__ gi,
                              const float* __restrict__ gh,
                              const float* __restrict__ zprev,
                              float* __restrict__ znext,
                              _Float16* __restrict__ z16)
{
  const size_t row = blockIdx.x;
  const int c = threadIdx.x;         // 256
  const float* gir = gi + row * 768;
  const float* ghr = gh + row * 768;
  const float r = 1.0f / (1.0f + __expf(-(gir[c]       + ghr[c])));
  const float z = 1.0f / (1.0f + __expf(-(gir[256 + c] + ghr[256 + c])));
  const float n = tanhf(gir[512 + c] + r * ghr[512 + c]);
  const float h = zprev[row * 256 + c];
  const float o = (1.0f - z) * n + z * h;
  znext[row * 256 + c] = o;
  z16[row * 256 + c] = (_Float16)o;
}

extern "C" void kernel_launch(void* const* d_in, const int* in_sizes, int n_in,
                              void* d_out, int out_size, void* d_ws, size_t ws_size,
                              hipStream_t stream)
{
  (void)in_sizes; (void)n_in; (void)out_size; (void)ws_size;
  typedef _Float16 H;

  const float* z_prev   = (const float*)d_in[0];
  const float* obs_loc  = (const float*)d_in[1];
  const float* act_loc  = (const float*)d_in[2];
  const float* obs_mask = (const float*)d_in[3];
  const float* id_feat  = (const float*)d_in[4];
  const int*   nidx     = (const int*)  d_in[5];
  const float* nmask    = (const float*)d_in[6];
  const float* edge     = (const float*)d_in[7];
  // params in insertion order: obs_enc(3x W,b), act_enc(2x), msg_enc(2x),
  // self_head(3x), neighbor_head(3x), gru(Wih,Whh,bih,bhh)
  const float* oW0=(const float*)d_in[8],  *ob0=(const float*)d_in[9];
  const float* oW1=(const float*)d_in[10], *ob1=(const float*)d_in[11];
  const float* oW2=(const float*)d_in[12], *ob2=(const float*)d_in[13];
  const float* aW0=(const float*)d_in[14], *ab0=(const float*)d_in[15];
  const float* aW1=(const float*)d_in[16], *ab1=(const float*)d_in[17];
  const float* mW0=(const float*)d_in[18], *mb0=(const float*)d_in[19];
  const float* mW1=(const float*)d_in[20], *mb1=(const float*)d_in[21];
  const float* sW0=(const float*)d_in[22], *sb0=(const float*)d_in[23];
  const float* sW1=(const float*)d_in[24], *sb1=(const float*)d_in[25];
  const float* sW2=(const float*)d_in[26], *sb2=(const float*)d_in[27];
  const float* nW0=(const float*)d_in[28], *nb0=(const float*)d_in[29];
  const float* nW1=(const float*)d_in[30], *nb1=(const float*)d_in[31];
  const float* nW2=(const float*)d_in[32], *nb2=(const float*)d_in[33];
  const float* Wih=(const float*)d_in[34];
  const float* Whh=(const float*)d_in[35];
  const float* bih=(const float*)d_in[36];
  const float* bhh=(const float*)d_in[37];

  const size_t NR = 8192;    // b*a rows
  const size_t MR = 131072;  // b*a*K msg rows

  // workspace layout (bytes); aliases exploit launch-order liveness
  char* ws = (char*)d_ws;
  const size_t OFF_UPD    = 0;                              // f16 8192x864
  const size_t OFF_Z16    = OFF_UPD    + NR * 864 * 2;      // f16 8192x256
  const size_t OFF_MSGIN  = OFF_Z16    + NR * 256 * 2;      // f16 131072x416
  const size_t OFF_MSGH1  = OFF_MSGIN  + MR * 416 * 2;      // f16 131072x512
  const size_t OFF_MSGOUT = OFF_MSGH1  + MR * 512 * 2;      // f16 131072x256
  const size_t OFF_WT     = OFF_MSGOUT + MR * 256 * 2;      // f16 transposed weights
  // aliases in MSGIN region (dead after msg layer 1):
  const size_t OFF_OBS1 = OFF_MSGIN;                        // f16 8192x512
  const size_t OFF_OBS2 = OFF_OBS1 + NR * 512 * 2;
  const size_t OFF_ACT1 = OFF_OBS2 + NR * 512 * 2;
  const size_t OFF_GI   = OFF_MSGIN;                        // f32 8192x768 (obs/act dead)
  const size_t OFF_GH   = OFF_GI + NR * 768 * 4;
  // aliases in MSGH1 region (dead after msg layer 2):
  const size_t OFF_SH1  = OFF_MSGH1;                        // f16 8192x512
  const size_t OFF_SH2  = OFF_SH1 + NR * 512 * 2;

  H* upd     = (H*)(ws + OFF_UPD);
  H* z16     = (H*)(ws + OFF_Z16);
  H* msg_in  = (H*)(ws + OFF_MSGIN);
  H* msg_h1  = (H*)(ws + OFF_MSGH1);
  H* msg_out = (H*)(ws + OFF_MSGOUT);
  H* obs1    = (H*)(ws + OFF_OBS1);
  H* obs2    = (H*)(ws + OFF_OBS2);
  H* act1    = (H*)(ws + OFF_ACT1);
  float* gi  = (float*)(ws + OFF_GI);
  float* gh  = (float*)(ws + OFF_GH);
  H* sh1     = (H*)(ws + OFF_SH1);
  H* sh2     = (H*)(ws + OFF_SH2);

  // transposed f16 weight buffers + f16 activation inputs (bump-allocated)
  size_t cur = OFF_WT;
  auto walloc = [&](size_t elems) { H* p = (H*)(ws + cur); cur += elems * 2; return p; };
  H* oW0t = walloc(512 *  128);  H* oW1t = walloc(512 * 512);  H* oW2t = walloc(256 * 512);
  H* aW0t = walloc(512 *   32);  H* aW1t = walloc(256 * 512);
  H* mW0t = walloc(512 *  416);  H* mW1t = walloc(256 * 512);
  H* sW0t = walloc(512 *  256);  H* sW1t = walloc(512 * 512);  H* sW2t = walloc(128 * 512);
  H* nW0t = walloc(512 *  256);  H* nW1t = walloc(512 * 512);  H* nW2t = walloc((size_t)4096 * 512);
  H* Wiht = walloc(768 *  864);  H* Whht = walloc(768 * 256);
  H* obs16 = walloc(NR * 128);
  H* act16 = walloc(NR * 32);
  H* zp16  = walloc(NR * 256);

  float* out       = (float*)d_out;
  float* z_next    = out;                                   // 8192x256
  float* self_pred = out + NR * 256;                        // 8192x128
  float* npred     = out + NR * 256 + NR * 128;             // 8192x4096

  const dim3 blk(256);

  // ---- weight transpose + f32->f16 (grid = (Nw/32, Kpad/32))
  transpose_w<<<dim3(16,  4), blk, 0, stream>>>(oW0, 128, 512, oW0t, 128);
  transpose_w<<<dim3(16, 16), blk, 0, stream>>>(oW1, 512, 512, oW1t, 512);
  transpose_w<<<dim3( 8, 16), blk, 0, stream>>>(oW2, 512, 256, oW2t, 512);
  transpose_w<<<dim3(16,  1), blk, 0, stream>>>(aW0,  32, 512, aW0t,  32);
  transpose_w<<<dim3( 8, 16), blk, 0, stream>>>(aW1, 512, 256, aW1t, 512);
  transpose_w<<<dim3(16, 13), blk, 0, stream>>>(mW0, 400, 512, mW0t, 416);
  transpose_w<<<dim3( 8, 16), blk, 0, stream>>>(mW1, 512, 256, mW1t, 512);
  transpose_w<<<dim3(16,  8), blk, 0, stream>>>(sW0, 256, 512, sW0t, 256);
  transpose_w<<<dim3(16, 16), blk, 0, stream>>>(sW1, 512, 512, sW1t, 512);
  transpose_w<<<dim3( 4, 16), blk, 0, stream>>>(sW2, 512, 128, sW2t, 512);
  transpose_w<<<dim3(16,  8), blk, 0, stream>>>(nW0, 256, 512, nW0t, 256);
  transpose_w<<<dim3(16, 16), blk, 0, stream>>>(nW1, 512, 512, nW1t, 512);
  transpose_w<<<dim3(128,16), blk, 0, stream>>>(nW2, 512, 4096, nW2t, 512);
  transpose_w<<<dim3(24, 27), blk, 0, stream>>>(Wih, 833, 768, Wiht, 864);
  transpose_w<<<dim3(24,  8), blk, 0, stream>>>(Whh, 256, 768, Whht, 256);

  // ---- f32 -> f16 activation inputs
  cvt_f16<<<dim3((NR * 128 + 255) / 256), blk, 0, stream>>>(obs_loc, obs16, NR * 128);
  cvt_f16<<<dim3((NR * 32  + 255) / 256), blk, 0, stream>>>(act_loc, act16, NR * 32);
  cvt_f16<<<dim3((NR * 256 + 255) / 256), blk, 0, stream>>>(z_prev,  zp16,  NR * 256);

  // ---- message branch (dominant: ~88 GFLOP)
  build_msg_in<<<dim3(131072), dim3(128), 0, stream>>>(z_prev, id_feat, nidx, edge, msg_in);
  gemm_wmma<H, true,  false><<<dim3(4, 1024), blk, 0, stream>>>(msg_in, 416, mW0t, mb0, msg_h1, 512, nullptr, 13);
  gemm_wmma<H, false, false><<<dim3(2, 1024), blk, 0, stream>>>(msg_h1, 512, mW1t, mb1, msg_out, 256, nullptr, 16);
  aggregate_msgs<<<dim3(8192), dim3(256), 0, stream>>>(msg_out, nmask, upd);

  // ---- obs encoder -> upd[:,0:256) (masked)
  gemm_wmma<H, true,  false><<<dim3(4, 64), blk, 0, stream>>>(obs16, 128, oW0t, ob0, obs1, 512, nullptr, 4);
  gemm_wmma<H, true,  false><<<dim3(4, 64), blk, 0, stream>>>(obs1, 512, oW1t, ob1, obs2, 512, nullptr, 16);
  gemm_wmma<H, false, true ><<<dim3(2, 64), blk, 0, stream>>>(obs2, 512, oW2t, ob2, upd + 0, 864, obs_mask, 16);

  // ---- act encoder -> upd[:,256:512) (masked)
  gemm_wmma<H, true,  false><<<dim3(4, 64), blk, 0, stream>>>(act16, 32, aW0t, ab0, act1, 512, nullptr, 1);
  gemm_wmma<H, false, true ><<<dim3(2, 64), blk, 0, stream>>>(act1, 512, aW1t, ab1, upd + 256, 864, obs_mask, 16);

  // ---- upd[:,768:864) = id | mask | zero-pad (K=833 padded to 864)
  fill_tail<<<dim3(8192), dim3(96), 0, stream>>>(id_feat, obs_mask, upd);

  // ---- GRU gates
  gemm_wmma<float, false, false><<<dim3(6, 64), blk, 0, stream>>>(upd, 864, Wiht, bih, gi, 768, nullptr, 27);
  gemm_wmma<float, false, false><<<dim3(6, 64), blk, 0, stream>>>(zp16, 256, Whht, bhh, gh, 768, nullptr, 8);
  gru_pointwise<<<dim3(8192), dim3(256), 0, stream>>>(gi, gh, z_prev, z_next, z16);

  // ---- self head -> self_pred (f32)
  gemm_wmma<H, true,  false><<<dim3(4, 64), blk, 0, stream>>>(z16, 256, sW0t, sb0, sh1, 512, nullptr, 8);
  gemm_wmma<H, true,  false><<<dim3(4, 64), blk, 0, stream>>>(sh1, 512, sW1t, sb1, sh2, 512, nullptr, 16);
  gemm_wmma<float, false, false><<<dim3(1, 64), blk, 0, stream>>>(sh2, 512, sW2t, sb2, self_pred, 128, nullptr, 16);

  // ---- neighbor head -> npred (f32, N=4096, ~41 GFLOP)
  gemm_wmma<H, true,  false><<<dim3(4, 64),  blk, 0, stream>>>(z16, 256, nW0t, nb0, sh1, 512, nullptr, 8);
  gemm_wmma<H, true,  false><<<dim3(4, 64),  blk, 0, stream>>>(sh1, 512, nW1t, nb1, sh2, 512, nullptr, 16);
  gemm_wmma<float, false, false><<<dim3(32, 64), blk, 0, stream>>>(sh2, 512, nW2t, nb2, npred, 4096, nullptr, 16);
}
